// Seq2SeqGRU_63410897158895
// MI455X (gfx1250) — matrix operand order repaired and checked by reading, not compiled, not run on hardware
//
#include <hip/hip_runtime.h>

// ---------------- problem constants ----------------
#define Bsz 512
#define Tlen 336
#define Ffeat 128
#define Hdim 512
#define NTgt 32
#define HORZ 48

typedef __attribute__((ext_vector_type(16))) _Float16 v16h;
typedef __attribute__((ext_vector_type(8)))  _Float16 v8h;
typedef __attribute__((ext_vector_type(8)))  float    v8f;

#define WMMA_F16(a, b, c) \
  __builtin_amdgcn_wmma_f32_16x16x32_f16(false, (a), false, (b), (short)0, (c), false, false)

#if defined(__HIP_DEVICE_COMPILE__) && __has_builtin(__builtin_amdgcn_tensor_load_to_lds)
#define HAVE_TDM 1
#else
#define HAVE_TDM 0
#endif

// ---- fragment loaders -------------------------------------------------
// A matrix 16x32 f16 (M x K), contiguous k order (same as load_b_frag, not the ISA table order):
//   lanes 0-15 : M=lane,    K = k0..k0+15
//   lanes 16-31: M=lane-16, K = k0+16..k0+31
__device__ __forceinline__ v16h load_a_frag(const _Float16* __restrict__ row, int k0, int hi) {
  const int ka = k0 + hi * 16;
  v8h lo = *(const v8h*)(row + ka);
  v8h hb = *(const v8h*)(row + ka + 8);
  v16h a;
#pragma unroll
  for (int i = 0; i < 8; ++i) { a[i] = lo[i]; a[8 + i] = hb[i]; }
  return a;
}

// B matrix 32x16 f16 (K x N): lane n (= lane&15) holds column n;
//   lanes 0-15 hold K=k0..k0+15, lanes 16-31 hold K=k0+16..k0+31.
__device__ __forceinline__ v16h load_b_frag(const _Float16* __restrict__ row, int k0, int hi) {
  const int kb = k0 + hi * 16;
  v8h lo = *(const v8h*)(row + kb);
  v8h hb = *(const v8h*)(row + kb + 8);
  v16h b;
#pragma unroll
  for (int i = 0; i < 8; ++i) { b[i] = lo[i]; b[8 + i] = hb[i]; }
  return b;
}

#if HAVE_TDM
typedef __attribute__((ext_vector_type(4))) unsigned int u32x4;
typedef __attribute__((ext_vector_type(8))) int          i32x8;
typedef __attribute__((ext_vector_type(4))) int          i32x4;

// Issue one TDM 2D tile load (rows x cols f16 elements, row stride in
// elements) from global memory into LDS at byte offset `ldsOff`.
// D# layout per CDNA5 ISA (groups 0/1; groups 2/3 zero).
// This toolchain's builtin takes 6 args: (g0, g1, g2, g3, g4, cpol).
__device__ __forceinline__ void tdm_load_2d(unsigned ldsOff, const void* gptr,
                                            unsigned rows, unsigned cols,
                                            unsigned tensorRows, unsigned strideElems) {
  unsigned long long ga = (unsigned long long)(uintptr_t)gptr;
  u32x4 g0;
  g0[0] = 1u;                                                  // count=1 (valid user D#)
  g0[1] = ldsOff;                                              // lds_addr (bytes)
  g0[2] = (unsigned)ga;                                        // global_addr[31:0]
  g0[3] = (unsigned)((ga >> 32) & 0x1FFFFFFu) | (2u << 30);    // addr[56:32] | type=2
  i32x8 g1;
  g1[0] = (int)(1u << 16);                                     // data_size=1 (2B), mask=0
  g1[1] = (int)((cols & 0xFFFFu) << 16);                       // tensor_dim0[15:0]
  g1[2] = (int)(((cols >> 16) & 0xFFFFu) | ((tensorRows & 0xFFFFu) << 16));
  g1[3] = (int)(((tensorRows >> 16) & 0xFFFFu) | ((cols & 0xFFFFu) << 16)); // tile_dim0
  g1[4] = (int)(rows & 0xFFFFu);                               // tile_dim1 ; tile_dim2=0
  g1[5] = (int)strideElems;                                    // tensor_dim0_stride lo32
  g1[6] = 0;                                                   // stride0 hi / stride1 lo
  g1[7] = 0;
  i32x4 gz4 = {0, 0, 0, 0};                                    // D# groups 2/3 (unused, 2D)
  i32x8 gz8 = {0, 0, 0, 0, 0, 0, 0, 0};
  __builtin_amdgcn_tensor_load_to_lds(g0, g1, gz4, gz4, gz8, 0);
}
#endif

// ---------------- prep kernels ----------------
__global__ void cvt_f32_to_f16_kernel(const float* __restrict__ src,
                                      _Float16* __restrict__ dst, int n) {
  int i = blockIdx.x * 256 + threadIdx.x;
  if (i < n) dst[i] = (_Float16)src[i];
}

__global__ void zero_f16_kernel(_Float16* __restrict__ p, int n) {
  int i = blockIdx.x * 256 + threadIdx.x;
  if (i < n) p[i] = (_Float16)0.0f;
}

// ---- software-pipelined triple-gate GEMM K-loop (compile-time K) ----
// Constant trip count => base + immediate-offset loads, and the next
// iteration's three B fragments prefetch into distinct registers so the
// v_wmma issue is not serialized behind s_wait_loadcnt 0.
template <int K>
__device__ __forceinline__ void gemm3_pipelined(const _Float16* __restrict__ aRow,
                                                const _Float16* __restrict__ wr,
                                                const _Float16* __restrict__ wz,
                                                const _Float16* __restrict__ wn,
                                                int hi,
                                                v8f& aR, v8f& aZ, v8f& aN) {
  v16h br = load_b_frag(wr, 0, hi);
  v16h bz = load_b_frag(wz, 0, hi);
  v16h bn = load_b_frag(wn, 0, hi);
#pragma unroll
  for (int k0 = 0; k0 < K; k0 += 32) {
    v16h a = load_a_frag(aRow, k0, hi);
    const int kn = (k0 + 32 < K) ? (k0 + 32) : k0;   // folds to a constant per iter
    v16h br2 = load_b_frag(wr, kn, hi);
    v16h bz2 = load_b_frag(wz, kn, hi);
    v16h bn2 = load_b_frag(wn, kn, hi);
    aR = WMMA_F16(a, br, aR);
    aZ = WMMA_F16(a, bz, aZ);
    aN = WMMA_F16(a, bn, aN);
    br = br2; bz = bz2; bn = bn2;
  }
}

// ---------------- fused GRU step (one layer, one timestep) ----------------
// Grid: (Hdim/32, Bsz/32).  Block: 128 threads = 4 waves.
// Wave w computes one 16x16 tile of h' (rowTile = w>>1, colTile = w&1),
// accumulating r/z/n gate GEMMs with WMMA, then applies the GRU update.
template <int INK, bool XF32>
__global__ void __launch_bounds__(128, 1)
gru_step_kernel(const void* __restrict__ xv, int xStride,
                const _Float16* __restrict__ h_in,
                _Float16* __restrict__ h_out,
                const _Float16* __restrict__ Wih,   // [3H, INK] f16 row-major
                const _Float16* __restrict__ Whh,   // [3H, H]   f16 row-major
                const float* __restrict__ bih,      // [3H] f32
                const float* __restrict__ bhh)      // [3H] f32
{
  extern __shared__ _Float16 smem[];
  _Float16* shH = smem;              // [32][Hdim]
  _Float16* shX = smem + 32 * Hdim;  // [32][INK]

  const int tid  = threadIdx.x;
  const int wave = tid >> 5;
  const int lane = tid & 31;
  const int rowBaseWG = blockIdx.y * 32;

#if HAVE_TDM
  // ---- TDM staging: DMA the h tile (and f16 x tile) straight into LDS ----
  if (wave == 0) {
    tdm_load_2d((unsigned)(uintptr_t)(void*)shH,
                h_in + (size_t)rowBaseWG * Hdim, 32, Hdim, Bsz, Hdim);
    if (!XF32) {
      tdm_load_2d((unsigned)(uintptr_t)(void*)shX,
                  (const _Float16*)xv + (size_t)rowBaseWG * xStride,
                  32, INK, Bsz, xStride);
    }
  }
  if (XF32) {             // encoder layer-0 raw input: convert f32->f16 manually
    const float* xs = (const float*)xv;
    const int cpr = INK / 4;
    for (int c = tid; c < 32 * cpr; c += 128) {
      int r = c / cpr, cc = c % cpr;
      float4 v = *(const float4*)(xs + (size_t)(rowBaseWG + r) * xStride + cc * 4);
      _Float16* d = shX + r * INK + cc * 4;
      d[0] = (_Float16)v.x; d[1] = (_Float16)v.y;
      d[2] = (_Float16)v.z; d[3] = (_Float16)v.w;
    }
  }
  if (wave == 0) __builtin_amdgcn_s_wait_tensorcnt(0);
  __syncthreads();
#else
  // ---- fallback cooperative staging (16B coalesced copies) ----
  {
    const int nch = 32 * (Hdim / 8);
    for (int c = tid; c < nch; c += 128) {
      int r = c >> 6, cc = c & 63;
      *(float4*)(shH + r * Hdim + cc * 8) =
          *(const float4*)(h_in + (size_t)(rowBaseWG + r) * Hdim + cc * 8);
    }
  }
  if (XF32) {
    const float* xs = (const float*)xv;
    const int cpr = INK / 4;
    for (int c = tid; c < 32 * cpr; c += 128) {
      int r = c / cpr, cc = c % cpr;
      float4 v = *(const float4*)(xs + (size_t)(rowBaseWG + r) * xStride + cc * 4);
      _Float16* d = shX + r * INK + cc * 4;
      d[0] = (_Float16)v.x; d[1] = (_Float16)v.y;
      d[2] = (_Float16)v.z; d[3] = (_Float16)v.w;
    }
  } else {
    const _Float16* xs = (const _Float16*)xv;
    const int cpr = INK / 8;
    for (int c = tid; c < 32 * cpr; c += 128) {
      int r = c / cpr, cc = c % cpr;
      *(float4*)(shX + r * INK + cc * 8) =
          *(const float4*)(xs + (size_t)(rowBaseWG + r) * xStride + cc * 8);
    }
  }
  __syncthreads();
#endif

  const int rowTile = wave >> 1;                       // 0..1
  const int colTile = wave & 1;                        // 0..1
  const int colBase = blockIdx.x * 32 + colTile * 16;  // h-unit base
  const int m  = lane & 15;
  const int hi = lane >> 4;

  v8f accR = {}, accZ = {}, accNi = {}, accNh = {};

  const _Float16* aRowX = shX + (rowTile * 16 + m) * INK;
  const _Float16* aRowH = shH + (rowTile * 16 + m) * Hdim;

  const _Float16* wih_r = Wih + (size_t)(0 * Hdim + colBase + m) * INK;
  const _Float16* wih_z = Wih + (size_t)(1 * Hdim + colBase + m) * INK;
  const _Float16* wih_n = Wih + (size_t)(2 * Hdim + colBase + m) * INK;
  const _Float16* whh_r = Whh + (size_t)(0 * Hdim + colBase + m) * Hdim;
  const _Float16* whh_z = Whh + (size_t)(1 * Hdim + colBase + m) * Hdim;
  const _Float16* whh_n = Whh + (size_t)(2 * Hdim + colBase + m) * Hdim;

  // warm the WGP cache for the (L2-resident) weight rows
  __builtin_prefetch(wih_r, 0, 3); __builtin_prefetch(wih_z, 0, 3);
  __builtin_prefetch(wih_n, 0, 3); __builtin_prefetch(whh_r, 0, 3);
  __builtin_prefetch(whh_z, 0, 3); __builtin_prefetch(whh_n, 0, 3);

  // gi = x @ Wih^T (K = INK) ; gh = h @ Whh^T (K = Hdim)
  gemm3_pipelined<INK>(aRowX, wih_r, wih_z, wih_n, hi, accR, accZ, accNi);
  gemm3_pipelined<Hdim>(aRowH, whh_r, whh_z, whh_n, hi, accR, accZ, accNh);

  // ---- GRU gate math + state update ----
  // C/D layout: VGPR v, lanes 0-15 -> M=v ; lanes 16-31 -> M=v+8 ; N = lane&15
  const int col = colBase + m;
  const float br_i = bih[col],            br_h = bhh[col];
  const float bz_i = bih[Hdim + col],     bz_h = bhh[Hdim + col];
  const float bn_i = bih[2 * Hdim + col], bn_h = bhh[2 * Hdim + col];

#pragma unroll
  for (int v = 0; v < 8; ++v) {
    const int rowLocal = rowTile * 16 + (hi ? v + 8 : v);
    float r = 1.0f / (1.0f + expf(-(accR[v] + br_i + br_h)));
    float z = 1.0f / (1.0f + expf(-(accZ[v] + bz_i + bz_h)));
    float n = tanhf(accNi[v] + bn_i + r * (accNh[v] + bn_h));
    float hold = (float)shH[rowLocal * Hdim + col];
    float hnew = (1.0f - z) * n + z * hold;
    h_out[(size_t)(rowBaseWG + rowLocal) * Hdim + col] = (_Float16)hnew;
  }
}

// ---------------- FC head: pred = h1 @ fcW^T + b ----------------
// Grid: Bsz/16 blocks, 64 threads (2 waves). Wave = output col tile (NT=32).
__global__ void __launch_bounds__(64, 1)
fc_kernel(const _Float16* __restrict__ h1,   // [B, H] f16
          const _Float16* __restrict__ fcW,  // [NT, H] f16
          const float* __restrict__ fcb,     // [NT]
          float* __restrict__ out,           // [B, HOR, NT] f32
          _Float16* __restrict__ dec_in,     // [B, NT] f16 feedback
          int tstep)
{
  const int tid = threadIdx.x;
  const int wave = tid >> 5, lane = tid & 31;
  const int m = lane & 15, hi = lane >> 4;
  const int rowBase = blockIdx.x * 16;
  const int colBase = wave * 16;

  v8f acc = {};
  const _Float16* aRow = h1  + (size_t)(rowBase + m) * Hdim;
  const _Float16* bRow = fcW + (size_t)(colBase + m) * Hdim;

  v16h b = load_b_frag(bRow, 0, hi);
#pragma unroll
  for (int k0 = 0; k0 < Hdim; k0 += 32) {
    v16h a  = load_a_frag(aRow, k0, hi);
    const int kn = (k0 + 32 < Hdim) ? (k0 + 32) : k0;
    v16h b2 = load_b_frag(bRow, kn, hi);
    acc = WMMA_F16(a, b, acc);
    b = b2;
  }

  const int nt = colBase + m;
  const float bias = fcb[nt];
#pragma unroll
  for (int v = 0; v < 8; ++v) {
    const int row = rowBase + (hi ? v + 8 : v);
    float p = acc[v] + bias;
    out[(size_t)row * HORZ * NTgt + (size_t)tstep * NTgt + nt] = p;
    dec_in[(size_t)row * NTgt + nt] = (_Float16)p;
  }
}

// ---------------- host driver ----------------
extern "C" void kernel_launch(void* const* d_in, const int* in_sizes, int n_in,
                              void* d_out, int out_size, void* d_ws, size_t ws_size,
                              hipStream_t stream) {
  (void)in_sizes; (void)n_in; (void)out_size; (void)ws_size;

  const float* x        = (const float*)d_in[0];
  const float* eWih0f32 = (const float*)d_in[1];
  const float* eWhh0f32 = (const float*)d_in[2];
  const float* ebih0    = (const float*)d_in[3];
  const float* ebhh0    = (const float*)d_in[4];
  const float* eWih1f32 = (const float*)d_in[5];
  const float* eWhh1f32 = (const float*)d_in[6];
  const float* ebih1    = (const float*)d_in[7];
  const float* ebhh1    = (const float*)d_in[8];
  const float* dWih0f32 = (const float*)d_in[9];
  const float* dWhh0f32 = (const float*)d_in[10];
  const float* dbih0    = (const float*)d_in[11];
  const float* dbhh0    = (const float*)d_in[12];
  const float* dWih1f32 = (const float*)d_in[13];
  const float* dWhh1f32 = (const float*)d_in[14];
  const float* dbih1    = (const float*)d_in[15];
  const float* dbhh1    = (const float*)d_in[16];
  const float* fcWf32   = (const float*)d_in[17];
  const float* fcb      = (const float*)d_in[18];
  float* out = (float*)d_out;

  // ---- workspace carve-out (f16 buffers, 256B aligned) ----
  char* ws = (char*)d_ws;
  size_t off = 0;
  auto alloc16 = [&](size_t elems) -> _Float16* {
    _Float16* p = (_Float16*)(ws + off);
    off = (off + elems * sizeof(_Float16) + 255) & ~(size_t)255;
    return p;
  };
  _Float16* eWih0 = alloc16((size_t)3 * Hdim * Ffeat);
  _Float16* eWhh0 = alloc16((size_t)3 * Hdim * Hdim);
  _Float16* eWih1 = alloc16((size_t)3 * Hdim * Hdim);
  _Float16* eWhh1 = alloc16((size_t)3 * Hdim * Hdim);
  _Float16* dWih0 = alloc16((size_t)3 * Hdim * NTgt);
  _Float16* dWhh0 = alloc16((size_t)3 * Hdim * Hdim);
  _Float16* dWih1 = alloc16((size_t)3 * Hdim * Hdim);
  _Float16* dWhh1 = alloc16((size_t)3 * Hdim * Hdim);
  _Float16* fcW   = alloc16((size_t)NTgt * Hdim);
  _Float16* h0b[2] = { alloc16((size_t)Bsz * Hdim), alloc16((size_t)Bsz * Hdim) };
  _Float16* h1b[2] = { alloc16((size_t)Bsz * Hdim), alloc16((size_t)Bsz * Hdim) };
  _Float16* dec_in = alloc16((size_t)Bsz * NTgt);

  // ---- prep: weight conversion + state zeroing ----
  auto cvt = [&](const float* s, _Float16* d, int n) {
    cvt_f32_to_f16_kernel<<<(n + 255) / 256, 256, 0, stream>>>(s, d, n);
  };
  cvt(eWih0f32, eWih0, 3 * Hdim * Ffeat);
  cvt(eWhh0f32, eWhh0, 3 * Hdim * Hdim);
  cvt(eWih1f32, eWih1, 3 * Hdim * Hdim);
  cvt(eWhh1f32, eWhh1, 3 * Hdim * Hdim);
  cvt(dWih0f32, dWih0, 3 * Hdim * NTgt);
  cvt(dWhh0f32, dWhh0, 3 * Hdim * Hdim);
  cvt(dWih1f32, dWih1, 3 * Hdim * Hdim);
  cvt(dWhh1f32, dWhh1, 3 * Hdim * Hdim);
  cvt(fcWf32,   fcW,   NTgt * Hdim);
  zero_f16_kernel<<<(Bsz * Hdim + 255) / 256, 256, 0, stream>>>(h0b[0], Bsz * Hdim);
  zero_f16_kernel<<<(Bsz * Hdim + 255) / 256, 256, 0, stream>>>(h1b[0], Bsz * Hdim);
  zero_f16_kernel<<<(Bsz * NTgt + 255) / 256, 256, 0, stream>>>(dec_in, Bsz * NTgt);

  const dim3 grid(Hdim / 32, Bsz / 32);
  const dim3 block(128);
  const size_t ldsF  = (size_t)(32 * Hdim + 32 * Ffeat) * 2;  // enc layer0
  const size_t ldsH  = (size_t)(32 * Hdim + 32 * Hdim) * 2;   // H-input layers
  const size_t ldsNT = (size_t)(32 * Hdim + 32 * NTgt) * 2;   // dec layer0

  int p0 = 0, p1 = 0;

  // ---- encoder: 2-layer GRU over T steps ----
  for (int t = 0; t < Tlen; ++t) {
    gru_step_kernel<Ffeat, true><<<grid, block, ldsF, stream>>>(
        (const void*)(x + (size_t)t * Ffeat), Tlen * Ffeat,
        h0b[p0], h0b[1 - p0], eWih0, eWhh0, ebih0, ebhh0);
    gru_step_kernel<Hdim, false><<<grid, block, ldsH, stream>>>(
        (const void*)h0b[1 - p0], Hdim,
        h1b[p1], h1b[1 - p1], eWih1, eWhh1, ebih1, ebhh1);
    p0 ^= 1; p1 ^= 1;
  }

  // ---- decoder: autoregressive, 48 steps ----
  for (int t = 0; t < HORZ; ++t) {
    gru_step_kernel<NTgt, false><<<grid, block, ldsNT, stream>>>(
        (const void*)dec_in, NTgt,
        h0b[p0], h0b[1 - p0], dWih0, dWhh0, dbih0, dbhh0);
    gru_step_kernel<Hdim, false><<<grid, block, ldsH, stream>>>(
        (const void*)h0b[1 - p0], Hdim,
        h1b[p1], h1b[1 - p1], dWih1, dWhh1, dbih1, dbhh1);
    fc_kernel<<<dim3(Bsz / 16), dim3(64), 0, stream>>>(
        h1b[1 - p1], fcW, fcb, out, dec_in, t);
    p0 ^= 1; p1 ^= 1;
  }
}
